// CoLAModel_69200513073836
// MI455X (gfx1250) — matrix operand order repaired and checked by reading, not compiled, not run on hardware
//
#include <hip/hip_runtime.h>

typedef __attribute__((ext_vector_type(16))) _Float16 v16h;
typedef __attribute__((ext_vector_type(8)))  float    v8f;

#define DIN  512
#define DOUT 256
#define NBLK 8192   // graph blocks of S=8 nodes

__device__ __forceinline__ float wave_sum(float v) {
#pragma unroll
  for (int o = 16; o > 0; o >>= 1) v += __shfl_down(v, o, 32);
  return v;
}

__global__ __launch_bounds__(256, 1)
void cola_fused_kernel(const float* __restrict__ pos_x,
                       const float* __restrict__ neg_x,
                       const int*   __restrict__ pos_src,
                       const int*   __restrict__ pos_dst,
                       const float* __restrict__ pos_w,
                       const int*   __restrict__ neg_src,
                       const int*   __restrict__ neg_dst,
                       const float* __restrict__ neg_w,
                       const float* __restrict__ W_gcn,
                       const float* __restrict__ b_gcn,
                       const float* __restrict__ prelu_a,
                       const float* __restrict__ W_bil,
                       const float* __restrict__ b_bil,
                       float* __restrict__ out)
{
  // One workgroup handles a 16-row M-tile = graph blocks (2*bid, 2*bid+1).
  __shared__ float Adj[2][2][8][8];     // [branch][blk][dst][src] weighted adjacency
  __shared__ float poolP[2][DOUT];      // pos pool   per blk
  __shared__ float poolN[2][DOUT];      // neg pool   per blk
  __shared__ float anch [2][DOUT];      // pos anchor per blk (shared by neg scores)
  __shared__ float ss[6];               // sumsq: |poolP0|,|poolP1|,|anch0|,|anch1|,|poolN0|,|poolN1|
  __shared__ float acc[4];              // scores: pos0,pos1,neg0,neg1 (unnormalized)

  const int tid  = threadIdx.x;
  const int lane = tid & 31;
  const int wave = tid >> 5;
  const int hi   = (lane >> 4) & 1;     // half-wave: selects K-interleave + row block
  const int col  = lane & 15;
  const int bid  = blockIdx.x;
  const int row0 = bid << 4;

  ((float*)Adj)[tid] = 0.0f;            // 2*2*64 = 256 floats, one per thread
  if (tid < 6) ss[tid]  = 0.0f;
  if (tid < 4) acc[tid] = 0.0f;
  __syncthreads();

  // Build both branches' 8x8 block adjacencies: 2 branches * 2 blocks * 64 edges = 256 = blockDim
  {
    const int br   = tid >> 7;
    const int j    = tid & 127;
    const int bl   = j >> 6;
    const int gblk = 2 * bid + bl;
    const int e    = gblk * 64 + (j & 63);
    const int*   srcp = br ? neg_src : pos_src;
    const int*   dstp = br ? neg_dst : pos_dst;
    const float* wp   = br ? neg_w   : pos_w;
    const int s = srcp[e] - gblk * 8;
    const int d = dstp[e] - gblk * 8;
    atomicAdd(&Adj[br][bl][d][s], wp[e]);   // ds_add_f32
  }
  __syncthreads();

  const float pa  = prelu_a[0];
  const int   cb0 = wave * 32;          // each of 8 waves owns 32 output columns (2 N-tiles)
  const int   cb1 = cb0 + 16;

  for (int br = 0; br < 2; ++br) {
    const float* __restrict__ x = br ? neg_x : pos_x;

    // GEMM: 16x256 tile of (x @ W_gcn), K=512, via v_wmma_f32_16x16x32_f16
    v8f c0 = {}; v8f c1 = {};
    // A fragment (16-bit 16x32 layout): row = lane&15; lanes<16 take K {0..7,16..23}, lanes>=16 {8..15,24..31}
    const float* arow  = x + (size_t)(row0 + col) * DIN + (hi ? 8 : 0);
    // B fragment (16-bit 32x16 layout): col = lane&15; lanes<16 take K {0..7,16..23}, lanes>=16 {8..15,24..31}
    const float* wbase = W_gcn + (size_t)(hi ? 8 : 0) * DOUT + col;

#pragma unroll 4
    for (int k0 = 0; k0 < DIN; k0 += 32) {
      v16h a;
      {
        const float4 f0 = *(const float4*)(arow + k0);
        const float4 f1 = *(const float4*)(arow + k0 + 4);
        const float4 f2 = *(const float4*)(arow + k0 + 16);
        const float4 f3 = *(const float4*)(arow + k0 + 20);
        a[0]=(_Float16)f0.x;  a[1]=(_Float16)f0.y;  a[2]=(_Float16)f0.z;  a[3]=(_Float16)f0.w;
        a[4]=(_Float16)f1.x;  a[5]=(_Float16)f1.y;  a[6]=(_Float16)f1.z;  a[7]=(_Float16)f1.w;
        a[8]=(_Float16)f2.x;  a[9]=(_Float16)f2.y;  a[10]=(_Float16)f2.z; a[11]=(_Float16)f2.w;
        a[12]=(_Float16)f3.x; a[13]=(_Float16)f3.y; a[14]=(_Float16)f3.z; a[15]=(_Float16)f3.w;
      }
      v16h b0, b1;
      const float* wp = wbase + (size_t)k0 * DOUT;
#pragma unroll
      for (int e = 0; e < 16; ++e) {
        b0[e] = (_Float16)wp[((e & 7) + ((e >> 3) << 4)) * DOUT + cb0];
        b1[e] = (_Float16)wp[((e & 7) + ((e >> 3) << 4)) * DOUT + cb1];
      }
      c0 = __builtin_amdgcn_wmma_f32_16x16x32_f16(false, a, false, b0, (short)0, c0, false, false);
      c1 = __builtin_amdgcn_wmma_f32_16x16x32_f16(false, a, false, b1, (short)0, c1, false, false);
    }

    // Block-local aggregation agg = A_blk @ xw, PReLU, pooling: all lane-local.
    // C layout: lanes<16 hold rows 0..7 (block 2*bid), lanes>=16 rows 8..15 (block 2*bid+1).
    const float (* __restrict__ A8)[8] = Adj[br][hi];
    float (*poolArr)[DOUT] = br ? poolN : poolP;

#pragma unroll
    for (int t = 0; t < 2; ++t) {
      const int   n  = (t ? cb1 : cb0) + col;
      const float bn = b_gcn[n];
      float pool = 0.0f, anchor = 0.0f;
#pragma unroll
      for (int r = 0; r < 8; ++r) {
        float sacc = bn;
#pragma unroll
        for (int q = 0; q < 8; ++q)
          sacc += A8[r][q] * (t ? c1[q] : c0[q]);
        const float h = (sacc >= 0.0f) ? sacc : pa * sacc;
        if (r < 7) pool += h; else anchor = h;   // mean of rows 0..6, anchor = row 7
      }
      poolArr[hi][n] = pool * (1.0f / 7.0f);
      if (br == 0) anch[hi][n] = anchor;
    }
  }
  __syncthreads();

  // Sum-of-squares for L2 norms (wave reduce + LDS atomic)
  {
    const int n = tid;
    const float p0 = poolP[0][n], p1 = poolP[1][n];
    const float a0 = anch [0][n], a1 = anch [1][n];
    const float q0 = poolN[0][n], q1 = poolN[1][n];
    const float s0 = wave_sum(p0 * p0), s1 = wave_sum(p1 * p1);
    const float s2 = wave_sum(a0 * a0), s3 = wave_sum(a1 * a1);
    const float s4 = wave_sum(q0 * q0), s5 = wave_sum(q1 * q1);
    if (lane == 0) {
      atomicAdd(&ss[0], s0); atomicAdd(&ss[1], s1);
      atomicAdd(&ss[2], s2); atomicAdd(&ss[3], s3);
      atomicAdd(&ss[4], s4); atomicAdd(&ss[5], s5);
    }
  }
  __syncthreads();

  // Bilinear scores: v_i = sum_j W_bil[i,j]*anchor[j]; score = sum_i pool[i]*v_i
#pragma unroll
  for (int blk = 0; blk < 2; ++blk) {
    const float* wr = W_bil + (size_t)tid * DOUT;
    float v = 0.0f;
#pragma unroll 4
    for (int j = 0; j < DOUT; ++j) v += wr[j] * anch[blk][j];
    const float pp = wave_sum(poolP[blk][tid] * v);
    const float pn = wave_sum(poolN[blk][tid] * v);
    if (lane == 0) { atomicAdd(&acc[blk], pp); atomicAdd(&acc[2 + blk], pn); }
  }
  __syncthreads();

  if (tid < 2) {
    const int   blk = tid;
    const float np = fmaxf(sqrtf(ss[blk]),     1e-12f);
    const float na = fmaxf(sqrtf(ss[2 + blk]), 1e-12f);
    const float nn = fmaxf(sqrtf(ss[4 + blk]), 1e-12f);
    const float bb = b_bil[0];
    out[2 * bid + blk]        = acc[blk]     / (np * na) + bb;  // pos score
    out[NBLK + 2 * bid + blk] = acc[2 + blk] / (nn * na) + bb;  // neg score
  }
}

extern "C" void kernel_launch(void* const* d_in, const int* in_sizes, int n_in,
                              void* d_out, int out_size, void* d_ws, size_t ws_size,
                              hipStream_t stream) {
  const float* pos_x   = (const float*)d_in[0];
  const float* neg_x   = (const float*)d_in[1];
  const int*   pos_src = (const int*)  d_in[2];
  const int*   pos_dst = (const int*)  d_in[3];
  const float* pos_w   = (const float*)d_in[4];
  const int*   neg_src = (const int*)  d_in[5];
  const int*   neg_dst = (const int*)  d_in[6];
  const float* neg_w   = (const float*)d_in[7];
  const float* W_gcn   = (const float*)d_in[8];
  const float* b_gcn   = (const float*)d_in[9];
  const float* prelu_a = (const float*)d_in[10];
  const float* W_bil   = (const float*)d_in[11];
  const float* b_bil   = (const float*)d_in[12];
  float* out = (float*)d_out;

  dim3 grid(NBLK / 2), block(256);
  cola_fused_kernel<<<grid, block, 0, stream>>>(
      pos_x, neg_x, pos_src, pos_dst, pos_w,
      neg_src, neg_dst, neg_w,
      W_gcn, b_gcn, prelu_a, W_bil, b_bil, out);
}
